// PsiNetwork_89936615178991
// MI455X (gfx1250) — matrix operand order repaired and checked by reading, not compiled, not run on hardware
//
#include <hip/hip_runtime.h>
#include <cmath>

// ---------------- problem constants (from reference) ----------------
#define NN      65536
#define EE      262144
#define HIDD    256
#define EDD     64
#define LL      2
#define HH      8
#define DHEAD   32
#define FFD     1024
#define ASTART  8192

typedef unsigned short u16;
typedef unsigned int   u32;

typedef __bf16 bf16_t;
typedef bf16_t v16bf __attribute__((ext_vector_type(16)));
typedef float  v8f   __attribute__((ext_vector_type(8)));
typedef int    i32x4 __attribute__((ext_vector_type(4)));

#define AS1 __attribute__((address_space(1)))
#define AS3 __attribute__((address_space(3)))

#if __has_builtin(__builtin_amdgcn_global_load_async_to_lds_b128) && \
    __has_builtin(__builtin_amdgcn_s_wait_asynccnt)
#define HAVE_ASYNC_LDS 1
#else
#define HAVE_ASYNC_LDS 0
#endif

// ---------------- helpers ----------------
__device__ __forceinline__ u16 f2bf(float f) {
    u32 u = __builtin_bit_cast(u32, f);
    u32 r = (u + 0x7FFFu + ((u >> 16) & 1u)) >> 16;   // round-to-nearest-even
    return (u16)r;
}
__device__ __forceinline__ float bf2f(u16 h) {
    return __builtin_bit_cast(float, (u32)h << 16);
}
__device__ __forceinline__ float gelu_exact(float x) {
    return 0.5f * x * (1.0f + erff(x * 0.70710678118654752440f));
}
__device__ __forceinline__ void atomicMaxFloat(float* addr, float val) {
    if (val >= 0.0f) atomicMax((int*)addr, __builtin_bit_cast(int, val));
    else             atomicMin((u32*)addr, __builtin_bit_cast(u32, val));
}
// copy 16B global -> LDS (async gfx1250 path; sync b128 fallback otherwise)
__device__ __forceinline__ void cp_g2l_b128(const u16* g, u16* l) {
#if HAVE_ASYNC_LDS
    __builtin_amdgcn_global_load_async_to_lds_b128(
        (AS1 i32x4*)(AS1 void*)(void*)g,
        (AS3 i32x4*)(AS3 void*)l, 0, 0);
#else
    *(uint4*)l = *(const uint4*)g;
#endif
}
__device__ __forceinline__ void cp_g2l_fence() {
#if HAVE_ASYNC_LDS
    __builtin_amdgcn_s_wait_asynccnt(0);
#endif
}

// ---------------- tiny utility kernels ----------------
__global__ void fill_f32(float* p, float v, long n) {
    long i = (long)blockIdx.x * blockDim.x + threadIdx.x;
    if (i < n) p[i] = v;
}
__global__ void conv_f32_bf16(const float* __restrict__ in, u16* __restrict__ out, long n) {
    long i = (long)blockIdx.x * blockDim.x + threadIdx.x;
    if (i < n) out[i] = f2bf(in[i]);
}
// convert + transpose: in[K][N] (f32) -> out[N][K] (bf16)
__global__ void convT_f32_bf16(const float* __restrict__ in, u16* __restrict__ out,
                               int K, int N) {
    long i = (long)blockIdx.x * blockDim.x + threadIdx.x;
    if (i >= (long)K * N) return;
    int k = (int)(i / N), n = (int)(i % N);
    out[(size_t)n * K + k] = f2bf(in[i]);
}

// ---------------- WMMA GEMM ----------------
// C[M,N] = A_bf16[M,K] @ B  with B pre-transposed as Bt_bf16[N,K].
// Block = 256 threads = 8 waves; block tile 128x128; wave tile 32x64 (2x4 WMMA accums).
// A tile (128x32) and Bt tile (128x32) staged in LDS via async b128 copies per K-step;
// all fragment reads are contiguous 16B LDS loads (ds_load_b128).
__global__ void __launch_bounds__(256)
gemm_bf16(const u16* __restrict__ A, const u16* __restrict__ Bt,
          const float* __restrict__ bias, float* __restrict__ Cf, u16* __restrict__ Cb,
          int M, int Nn, int K, int flags /*1=bias,2=gelu*/) {
    __shared__ __align__(16) u16 As[128 * 32];   // 8KB
    __shared__ __align__(16) u16 Bs[128 * 32];   // 8KB

    const int tid   = threadIdx.x;
    const int lane  = tid & 31;
    const int wid   = tid >> 5;
    const int waveM = wid >> 1;          // 0..3  -> 32 rows each
    const int waveN = wid & 1;           // 0..1  -> 64 cols each
    const int tileM0 = blockIdx.y * 128;
    const int tileN0 = blockIdx.x * 128;

    // fragment lane geometry (ISA 7.12.2 layouts)
    const int l15   = lane & 15;
    const int hi8   = (lane >> 4) * 8;   // A K-subgroup select
    const int bk_hi = (lane >> 4) * 16;  // B K-subgroup select (used as 8h bf16 via >>2)

    v8f acc[2][4] = {};

    for (int kk = 0; kk < K; kk += 32) {
        // ---- stage A tile: 128 rows x 32 K bf16, 2 x b128 per thread ----
        #pragma unroll
        for (int i = 0; i < 2; ++i) {
            int lin = tid + i * 256;
            int row = lin >> 2;
            int cs  = (lin & 3) * 8;
            cp_g2l_b128(A + (size_t)(tileM0 + row) * K + kk + cs,
                        As + row * 32 + cs);
        }
        // ---- stage Bt tile: 128 N-rows x 32 K bf16, 2 x b128 per thread ----
        #pragma unroll
        for (int i = 0; i < 2; ++i) {
            int lin = tid + i * 256;
            int row = lin >> 2;
            int cs  = (lin & 3) * 8;
            cp_g2l_b128(Bt + (size_t)(tileN0 + row) * K + kk + cs,
                        Bs + row * 32 + cs);
        }
        cp_g2l_fence();
        __syncthreads();

        // ---- build fragments from LDS (all contiguous b128 reads) ----
        union { u32 u[8]; v16bf v; } fa[2], fb[4];
        #pragma unroll
        for (int mt = 0; mt < 2; ++mt) {
            int m = waveM * 32 + mt * 16 + l15;
            const u32* arow = (const u32*)(As + m * 32);   // 16 u32 per row
            #pragma unroll
            for (int g = 0; g < 2; ++g)
                #pragma unroll
                for (int p = 0; p < 4; ++p) {
                    int k0 = g * 16 + hi8 + p * 2;         // even
                    fa[mt].u[g * 4 + p] = arow[k0 >> 1];
                }
        }
        #pragma unroll
        for (int nt = 0; nt < 4; ++nt) {
            int n = waveN * 64 + nt * 16 + l15;
            const u32* brow = (const u32*)(Bs + n * 32);   // 16 u32 per row
            #pragma unroll
            for (int j = 0; j < 8; ++j)
                fb[nt].u[j] = brow[(j >> 2) * 8 + (bk_hi >> 2) + (j & 3)];      // K table order (matches A)
        }

        // ---- 8 WMMAs per K-step ----
        #pragma unroll
        for (int mt = 0; mt < 2; ++mt)
            #pragma unroll
            for (int nt = 0; nt < 4; ++nt)
                acc[mt][nt] = __builtin_amdgcn_wmma_f32_16x16x32_bf16(
                    false, fa[mt].v, false, fb[nt].v, (short)0, acc[mt][nt],
                    false, false);

        __syncthreads();
    }

    // ---- epilogue ----
    #pragma unroll
    for (int mt = 0; mt < 2; ++mt) {
        #pragma unroll
        for (int nt = 0; nt < 4; ++nt) {
            int col = tileN0 + waveN * 64 + nt * 16 + l15;
            float bv = (flags & 1) ? bias[col] : 0.0f;
            #pragma unroll
            for (int r = 0; r < 8; ++r) {
                int row = tileM0 + waveM * 32 + mt * 16 + r + ((lane >> 4) << 3);
                float v = acc[mt][nt][r] + bv;
                if (flags & 2) v = gelu_exact(v);
                if (Cf) Cf[(size_t)row * Nn + col] = v;
                if (Cb) Cb[(size_t)row * Nn + col] = f2bf(v);
            }
        }
    }
}

// ---------------- attention: per-edge score + segment max ----------------
__global__ void attn_score(const float* __restrict__ q, const float* __restrict__ k,
                           const float* __restrict__ ea, const u16* __restrict__ WeKb,
                           const int* __restrict__ src, const int* __restrict__ dst,
                           float* __restrict__ score, float* __restrict__ smax, int E) {
    __shared__ u16 W[EDD * HIDD];
    for (int i = threadIdx.x; i < (EDD * HIDD) / 2; i += blockDim.x)
        ((u32*)W)[i] = ((const u32*)WeKb)[i];
    __syncthreads();

    int e = blockIdx.x * (blockDim.x >> 5) + (threadIdx.x >> 5);
    if (e >= E) return;
    int lane = threadIdx.x & 31;
    int s = src[e], d = dst[e];
    int base = lane * 8;

    float qv[8], kv[8], ek[8];
    const float* qr = q + (size_t)d * HIDD + base;
    const float* kr = k + (size_t)s * HIDD + base;
    #pragma unroll
    for (int i = 0; i < 8; ++i) { qv[i] = qr[i]; kv[i] = kr[i]; ek[i] = 0.0f; }

    const float* ear = ea + (size_t)e * EDD;
    for (int c = 0; c < EDD; ++c) {
        float a = ear[c];
        const u16* wr = &W[c * HIDD + base];
        #pragma unroll
        for (int i = 0; i < 8; ++i) ek[i] += a * bf2f(wr[i]);
    }
    float sc = 0.0f;
    #pragma unroll
    for (int i = 0; i < 8; ++i) sc += qv[i] * (kv[i] + ek[i]);
    sc += __shfl_xor(sc, 1, 32);
    sc += __shfl_xor(sc, 2, 32);
    sc *= 0.17677669529663687f;   // 1/sqrt(32)
    if ((lane & 3) == 0) {
        int h = lane >> 2;
        score[(size_t)e * HH + h] = sc;
        atomicMaxFloat(&smax[(size_t)d * HH + h], sc);
    }
}

// ---------------- attention: p = exp(score - smax[dst]); denom += p ----------------
__global__ void attn_p(const float* __restrict__ score, const float* __restrict__ smax,
                       const int* __restrict__ dst, float* __restrict__ p,
                       float* __restrict__ denom, int E) {
    int tid = blockIdx.x * blockDim.x + threadIdx.x;
    if (tid >= E * HH) return;
    int e = tid >> 3, h = tid & 7;
    int d = dst[e];
    float pv = expf(score[tid] - smax[(size_t)d * HH + h]);
    p[tid] = pv;
    atomicAdd(&denom[(size_t)d * HH + h], pv);
}

// ---------------- attention: agg[dst] += alpha * (v[src] + ea@WeV) ----------------
__global__ void attn_agg(const float* __restrict__ p, const float* __restrict__ denom,
                         const float* __restrict__ v, const float* __restrict__ ea,
                         const u16* __restrict__ WeVb,
                         const int* __restrict__ src, const int* __restrict__ dst,
                         float* __restrict__ agg, int E) {
    __shared__ u16 W[EDD * HIDD];
    for (int i = threadIdx.x; i < (EDD * HIDD) / 2; i += blockDim.x)
        ((u32*)W)[i] = ((const u32*)WeVb)[i];
    __syncthreads();

    int e = blockIdx.x * (blockDim.x >> 5) + (threadIdx.x >> 5);
    if (e >= E) return;
    int lane = threadIdx.x & 31;
    int s = src[e], d = dst[e];
    int base = lane * 8;
    int h = lane >> 2;

    float alpha = p[(size_t)e * HH + h] / denom[(size_t)d * HH + h];

    float ev[8];
    #pragma unroll
    for (int i = 0; i < 8; ++i) ev[i] = 0.0f;
    const float* ear = ea + (size_t)e * EDD;
    for (int c = 0; c < EDD; ++c) {
        float a = ear[c];
        const u16* wr = &W[c * HIDD + base];
        #pragma unroll
        for (int i = 0; i < 8; ++i) ev[i] += a * bf2f(wr[i]);
    }
    const float* vr = v + (size_t)s * HIDD + base;
    float* ar = agg + (size_t)d * HIDD + base;
    #pragma unroll
    for (int i = 0; i < 8; ++i) atomicAdd(&ar[i], alpha * (vr[i] + ev[i]));
}

// ---------------- residual + LayerNorm (one wave per 256-dim row) ----------------
// writes f32 x and bf16 xb (bf16 feeds the next WMMA GEMM)
__global__ void add_ln(const float* __restrict__ res, const float* __restrict__ t,
                       const float* __restrict__ sc, const float* __restrict__ bi,
                       float* __restrict__ xout, u16* __restrict__ xbout, int nrows) {
    int row = blockIdx.x * (blockDim.x >> 5) + (threadIdx.x >> 5);
    if (row >= nrows) return;
    int lane = threadIdx.x & 31;
    int base = lane * 8;
    const float* rr = res + (size_t)row * HIDD + base;
    const float* tr = t   + (size_t)row * HIDD + base;
    float y[8]; float sum = 0.0f, sq = 0.0f;
    #pragma unroll
    for (int i = 0; i < 8; ++i) { y[i] = rr[i] + tr[i]; sum += y[i]; sq += y[i] * y[i]; }
    #pragma unroll
    for (int m = 1; m < 32; m <<= 1) {
        sum += __shfl_xor(sum, m, 32);
        sq  += __shfl_xor(sq,  m, 32);
    }
    float mu  = sum * (1.0f / HIDD);
    float var = sq * (1.0f / HIDD) - mu * mu;
    float inv = rsqrtf(var + 1e-5f);
    float* xo = xout + (size_t)row * HIDD + base;
    u16*   xb = xbout + (size_t)row * HIDD + base;
    #pragma unroll
    for (int i = 0; i < 8; ++i) {
        float o = (y[i] - mu) * inv * sc[base + i] + bi[base + i];
        xo[i] = o;
        xb[i] = f2bf(o);
    }
}

// ---------------- head: out[m,7] = t[m,:] @ w2 + b2 ----------------
__global__ void head2_k(const float* __restrict__ t, const float* __restrict__ w,
                        const float* __restrict__ b, float* __restrict__ out, int M) {
    int tid = blockIdx.x * blockDim.x + threadIdx.x;
    if (tid >= M * 7) return;
    int m = tid / 7, j = tid % 7;
    const float* tr = t + (size_t)m * HIDD;
    float acc = b[j];
    for (int c = 0; c < HIDD; ++c) acc += tr[c] * w[c * 7 + j];
    out[tid] = acc;
}

// ---------------- host orchestration ----------------
static inline void launch_gemm(const u16* A, const u16* Bt, const float* bias,
                               float* Cf, u16* Cb, int M, int Nn, int K, int flags,
                               hipStream_t s) {
    dim3 grid(Nn / 128, M / 128);
    gemm_bf16<<<grid, 256, 0, s>>>(A, Bt, bias, Cf, Cb, M, Nn, K, flags);
}
static inline void launch_fill(float* p, float v, long n, hipStream_t s) {
    fill_f32<<<(int)((n + 255) / 256), 256, 0, s>>>(p, v, n);
}
static inline void launch_conv(const float* in, u16* out, long n, hipStream_t s) {
    conv_f32_bf16<<<(int)((n + 255) / 256), 256, 0, s>>>(in, out, n);
}
static inline void launch_convT(const float* in, u16* out, int K, int N, hipStream_t s) {
    long n = (long)K * N;
    convT_f32_bf16<<<(int)((n + 255) / 256), 256, 0, s>>>(in, out, K, N);
}

extern "C" void kernel_launch(void* const* d_in, const int* in_sizes, int n_in,
                              void* d_out, int out_size, void* d_ws, size_t ws_size,
                              hipStream_t stream) {
    const float* x_in      = (const float*)d_in[0];
    const float* edge_attr = (const float*)d_in[1];
    const float* Wq        = (const float*)d_in[2];
    const float* Wk        = (const float*)d_in[3];
    const float* Wv        = (const float*)d_in[4];
    const float* WeK       = (const float*)d_in[5];
    const float* WeV       = (const float*)d_in[6];
    const float* Wo        = (const float*)d_in[7];
    const float* ln1_s     = (const float*)d_in[8];
    const float* ln1_b     = (const float*)d_in[9];
    const float* ffn_w1    = (const float*)d_in[10];
    const float* ffn_b1    = (const float*)d_in[11];
    const float* ffn_w2    = (const float*)d_in[12];
    const float* ffn_b2    = (const float*)d_in[13];
    const float* ln2_s     = (const float*)d_in[14];
    const float* ln2_b     = (const float*)d_in[15];
    const float* head_w1   = (const float*)d_in[16];
    const float* head_b1   = (const float*)d_in[17];
    const float* head_w2   = (const float*)d_in[18];
    const float* head_b2   = (const float*)d_in[19];
    const int*   eidx      = (const int*)d_in[20];
    const int*   src       = eidx;
    const int*   dst       = eidx + EE;
    float*       out       = (float*)d_out;

    // ---- carve workspace ----
    char*  ws  = (char*)d_ws;
    size_t off = 0;
    auto allocF = [&](size_t n) { float* p = (float*)(ws + off); off += n * 4; return p; };
    auto allocU = [&](size_t n) { u16*   p = (u16*)  (ws + off); off += n * 2; return p; };

    float* x     = allocF((size_t)NN * HIDD);
    float* q     = allocF((size_t)NN * HIDD);
    float* k     = allocF((size_t)NN * HIDD);
    float* v     = allocF((size_t)NN * HIDD);
    float* agg   = allocF((size_t)NN * HIDD);
    float* tmp   = allocF((size_t)NN * HIDD);
    float* score = allocF((size_t)EE * HH);
    float* pbuf  = allocF((size_t)EE * HH);
    float* smax  = allocF((size_t)NN * HH);
    float* denom = allocF((size_t)NN * HH);

    u16* xb   = allocU((size_t)NN * HIDD);
    u16* aggb = allocU((size_t)NN * HIDD);
    u16* hidb = allocU((size_t)NN * FFD);

    // transposed bf16 weights [N][K] for WMMA GEMMs
    u16* WqT  = allocU((size_t)LL * HIDD * HIDD);
    u16* WkT  = allocU((size_t)LL * HIDD * HIDD);
    u16* WvT  = allocU((size_t)LL * HIDD * HIDD);
    u16* WoT  = allocU((size_t)LL * HIDD * HIDD);
    u16* W1T  = allocU((size_t)LL * HIDD * FFD);
    u16* W2T  = allocU((size_t)LL * FFD * HIDD);
    u16* HW1T = allocU((size_t)HIDD * HIDD);
    // row-major bf16 edge-projection weights (used by attention LDS kernels)
    u16* WeKb = allocU((size_t)LL * EDD * HIDD);
    u16* WeVb = allocU((size_t)LL * EDD * HIDD);

    // ---- weight conversions (transpose for GEMM operands) ----
    for (int l = 0; l < LL; ++l) {
        size_t o256 = (size_t)l * HIDD * HIDD;
        size_t oFF  = (size_t)l * HIDD * FFD;
        launch_convT(Wq + o256, WqT + o256, HIDD, HIDD, stream);
        launch_convT(Wk + o256, WkT + o256, HIDD, HIDD, stream);
        launch_convT(Wv + o256, WvT + o256, HIDD, HIDD, stream);
        launch_convT(Wo + o256, WoT + o256, HIDD, HIDD, stream);
        launch_convT(ffn_w1 + oFF, W1T + oFF, HIDD, FFD, stream);
        launch_convT(ffn_w2 + oFF, W2T + oFF, FFD, HIDD, stream);
    }
    launch_convT(head_w1, HW1T, HIDD, HIDD, stream);
    launch_conv(WeK, WeKb, (long)LL * EDD * HIDD, stream);
    launch_conv(WeV, WeVb, (long)LL * EDD * HIDD, stream);

    // mutable f32 copy + bf16 copy of node features
    (void)hipMemcpyAsync(x, x_in, (size_t)NN * HIDD * 4, hipMemcpyDeviceToDevice, stream);
    launch_conv(x_in, xb, (long)NN * HIDD, stream);

    const float NEG_INF = -__builtin_huge_valf();

    for (int l = 0; l < LL; ++l) {
        const u16* WqT_l = WqT + (size_t)l * HIDD * HIDD;
        const u16* WkT_l = WkT + (size_t)l * HIDD * HIDD;
        const u16* WvT_l = WvT + (size_t)l * HIDD * HIDD;
        const u16* WoT_l = WoT + (size_t)l * HIDD * HIDD;
        const u16* W1T_l = W1T + (size_t)l * HIDD * FFD;
        const u16* W2T_l = W2T + (size_t)l * FFD * HIDD;
        const u16* WeK_l = WeKb + (size_t)l * EDD * HIDD;
        const u16* WeV_l = WeVb + (size_t)l * EDD * HIDD;

        // q/k/v projections (WMMA bf16, LDS-staged, async copies)
        launch_gemm(xb, WqT_l, nullptr, q, nullptr, NN, HIDD, HIDD, 0, stream);
        launch_gemm(xb, WkT_l, nullptr, k, nullptr, NN, HIDD, HIDD, 0, stream);
        launch_gemm(xb, WvT_l, nullptr, v, nullptr, NN, HIDD, HIDD, 0, stream);

        // segment-softmax attention over edges
        launch_fill(smax, NEG_INF, (long)NN * HH, stream);
        launch_fill(denom, 0.0f,  (long)NN * HH, stream);
        launch_fill(agg,   0.0f,  (long)NN * HIDD, stream);

        attn_score<<<EE / 8, 256, 0, stream>>>(q, k, edge_attr, WeK_l, src, dst,
                                               score, smax, EE);
        attn_p<<<(EE * HH) / 256, 256, 0, stream>>>(score, smax, dst, pbuf, denom, EE);
        attn_agg<<<EE / 8, 256, 0, stream>>>(pbuf, denom, v, edge_attr, WeV_l,
                                             src, dst, agg, EE);

        // output projection + LN1
        launch_conv(agg, aggb, (long)NN * HIDD, stream);
        launch_gemm(aggb, WoT_l, nullptr, tmp, nullptr, NN, HIDD, HIDD, 0, stream);
        add_ln<<<NN / 8, 256, 0, stream>>>(x, tmp, ln1_s + (size_t)l * HIDD,
                                           ln1_b + (size_t)l * HIDD, x, xb, NN);

        // FFN + LN2  (hidden kept in bf16, written by GEMM epilogue)
        launch_gemm(xb, W1T_l, ffn_b1 + (size_t)l * FFD, nullptr, hidb,
                    NN, FFD, HIDD, 1 | 2, stream);
        launch_gemm(hidb, W2T_l, ffn_b2 + (size_t)l * HIDD, tmp, nullptr,
                    NN, HIDD, FFD, 1, stream);
        add_ln<<<NN / 8, 256, 0, stream>>>(x, tmp, ln2_s + (size_t)l * HIDD,
                                           ln2_b + (size_t)l * HIDD, x, xb, NN);
    }

    // head: gelu(a @ head_w1 + b1) @ head_w2 + b2
    const int Ma = NN - ASTART;   // 57344, multiple of 128
    launch_gemm(xb + (size_t)ASTART * HIDD, HW1T, head_b1, tmp, nullptr,
                Ma, HIDD, HIDD, 1 | 2, stream);
    head2_k<<<(Ma * 7 + 255) / 256, 256, 0, stream>>>(tmp, head_w2, head_b2, out, Ma);
}
